// GroupQueryAttention_83846351553167
// MI455X (gfx1250) — matrix operand order repaired and checked by reading, not compiled, not run on hardware
//
#include <hip/hip_runtime.h>
#include <hip/hip_bf16.h>
#include <math.h>

// ---------------- problem constants ----------------
#define B_   2
#define S_   2048
#define DIM_ 2048
#define H_   32
#define KV_  8
#define D_   64
#define WINH 256          // WIN/2
#define NT_  (B_ * S_)    // 4096 token rows

typedef __attribute__((ext_vector_type(16))) __bf16 bf16x16;
typedef __attribute__((ext_vector_type(8)))  __bf16 bf16x8;
typedef __attribute__((ext_vector_type(8)))  float  f32x8;

union ABFrag {
    bf16x16 v;
    bf16x8  h[2];
    __bf16  e[16];
};

// ---------------- fp32 -> bf16 convert ----------------
__global__ void f32_to_bf16_kernel(const float* __restrict__ in,
                                   __bf16* __restrict__ out, int n) {
    int i = blockIdx.x * blockDim.x + threadIdx.x;
    if (i < n) out[i] = (__bf16)in[i];
}

// ---------------- register-blocked WMMA GEMM: C[N,M] = A[N,K] * B[M,K]^T ----
// A, B row-major bf16, K contiguous. One wave per 32(N) x 64(M) C macro-tile:
// 2 A frags x 4 B frags -> 8 independent v_wmma_f32_16x16x32_bf16 per K-step.
template <bool OUT_F32>
__global__ void gemm_nt_bf16_kernel(const __bf16* __restrict__ A,
                                    const __bf16* __restrict__ Bm,
                                    void* __restrict__ Cout,
                                    int N, int M, int Kd) {
    const int lane = threadIdx.x & 31;
    const int wave = threadIdx.x >> 5;
    const int tile = blockIdx.x * (blockDim.x >> 5) + wave;
    const int tiles_m = M >> 6;                   // 64-wide macro-tiles
    const int n0 = (tile / tiles_m) << 5;         // 32 rows
    const int m0 = (tile % tiles_m) << 6;         // 64 cols
    if (n0 >= N) return;

    const int half = lane >> 4;
    const int lidx = lane & 15;

    const __bf16* arow0 = A + (size_t)(n0 + lidx) * Kd;
    const __bf16* arow1 = A + (size_t)(n0 + 16 + lidx) * Kd;
    const __bf16* brow[4];
#pragma unroll
    for (int c = 0; c < 4; ++c)
        brow[c] = Bm + (size_t)(m0 + c * 16 + lidx) * Kd;

    f32x8 acc[2][4];
#pragma unroll
    for (int r = 0; r < 2; ++r)
#pragma unroll
        for (int c = 0; c < 4; ++c) acc[r][c] = (f32x8){};

    for (int k0 = 0; k0 < Kd; k0 += 32) {
        ABFrag a[2], b[4];
        a[0].h[0] = *(const bf16x8*)(arow0 + k0 + 16 * half);
        a[0].h[1] = *(const bf16x8*)(arow0 + k0 + 16 * half + 8);
        a[1].h[0] = *(const bf16x8*)(arow1 + k0 + 16 * half);
        a[1].h[1] = *(const bf16x8*)(arow1 + k0 + 16 * half + 8);
#pragma unroll
        for (int c = 0; c < 4; ++c) {
            b[c].h[0] = *(const bf16x8*)(brow[c] + k0 + 16 * half);
            b[c].h[1] = *(const bf16x8*)(brow[c] + k0 + 16 * half + 8);
        }
#pragma unroll
        for (int r = 0; r < 2; ++r)
#pragma unroll
            for (int c = 0; c < 4; ++c)
                acc[r][c] = __builtin_amdgcn_wmma_f32_16x16x32_bf16(
                    false, a[r].v, false, b[c].v, (short)0, acc[r][c],
                    false, false);
    }

#pragma unroll
    for (int r = 0; r < 2; ++r)
#pragma unroll
        for (int c = 0; c < 4; ++c)
#pragma unroll
            for (int v = 0; v < 8; ++v) {
                size_t idx = (size_t)(n0 + r * 16 + v + 8 * half) * M
                           + m0 + c * 16 + lidx;
                if (OUT_F32) ((float*)Cout)[idx] = acc[r][c][v];
                else         ((__bf16*)Cout)[idx] = (__bf16)acc[r][c][v];
            }
}

// ---------------- RoPE (in-place on bf16 [B*S, heads, 64]) ----------------
__global__ void rope_kernel(__bf16* __restrict__ t, int heads, int total) {
    int idx = blockIdx.x * blockDim.x + threadIdx.x;
    if (idx >= total) return;
    int d2  = idx & 31;                 // 0..31
    int hh  = (idx >> 5) % heads;
    int tok = (idx >> 5) / heads;       // b*S + s
    int pos = tok & (S_ - 1);

    float inv = __powf(10000.0f, -(float)d2 / 32.0f);
    float ang = (float)pos * inv;
    float sv, cv;
    __sincosf(ang, &sv, &cv);

    __bf16* base = t + ((size_t)tok * heads + hh) * D_ + d2;
    float lo = (float)base[0];
    float hi = (float)base[32];
    base[0]  = (__bf16)(lo * cv - hi * sv);
    base[32] = (__bf16)(hi * cv + lo * sv);
}

// ---------------- flash sliding-window GQA attention ----------------
// 256-thread blocks, 8 waves; each wave owns one 16-query tile and a private
// 16x32 LDS P-slice. Key loop steps 32 keys: 2 score tiles (4 WMMA) + full-K
// P*V (4 WMMA, no zero padding). Same-wave LDS ordering via s_wait_dscnt.
__global__ void __launch_bounds__(256)
attn_kernel(const __bf16* __restrict__ qb,  // [B*S, H, 64]
            const __bf16* __restrict__ kb,  // [B*S, KV, 64]
            const __bf16* __restrict__ vb,  // [B*S, KV, 64]
            __bf16* __restrict__ ob) {      // [B*S, H, 64]
    const int lane = threadIdx.x & 31;
    const int wave = threadIdx.x >> 5;
    const int half = lane >> 4;
    const int lidx = lane & 15;
    const int qt = blockIdx.x * 8 + wave;   // 0 .. S/16-1
    const int h  = blockIdx.y;
    const int b  = blockIdx.z;
    const int kvh = h >> 2;                 // H/KV = 4
    const int i0 = qt << 4;
    const float scale = 0.125f;             // 1/sqrt(64)

    __shared__ __bf16 lds_p[8 * 16 * 32];   // per-wave 16x32 slice
    __bf16* pslice = lds_p + wave * (16 * 32);

    // Q fragments: A layout, row = i0+lidx, frag c covers d = 32c..32c+31
    ABFrag qa[2];
    {
        const __bf16* qrow = qb + (((size_t)b * S_ + i0 + lidx) * H_ + h) * D_;
#pragma unroll
        for (int c = 0; c < 2; ++c) {
            qa[c].h[0] = *(const bf16x8*)(qrow + 32 * c + 16 * half);
            qa[c].h[1] = *(const bf16x8*)(qrow + 32 * c + 16 * half + 8);
        }
    }

    float m_i[8], l_i[8];
    f32x8 o_acc[4];
#pragma unroll
    for (int v = 0; v < 8; ++v) { m_i[v] = -INFINITY; l_i[v] = 0.0f; }
#pragma unroll
    for (int c = 0; c < 4; ++c) o_acc[c] = (f32x8){};

    const int j_lo = (i0 >= WINH) ? ((i0 - WINH) & ~31) : 0;  // 32-aligned
    const int j_hi = i0 + 15;                                 // last valid key

    for (int j0 = j_lo; j0 <= j_hi; j0 += 32) {
        // K^T fragments for two 16-key sub-tiles: B layout, col = key
        ABFrag kf[2][2];   // [subtile t][d-frag c]
#pragma unroll
        for (int t = 0; t < 2; ++t) {
            const __bf16* krow =
                kb + (((size_t)b * S_ + j0 + 16 * t + lidx) * KV_ + kvh) * D_;
#pragma unroll
            for (int c = 0; c < 2; ++c) {
                kf[t][c].h[0] = *(const bf16x8*)(krow + 32 * c + 16 * half);
                kf[t][c].h[1] = *(const bf16x8*)(krow + 32 * c + 16 * half + 8);
            }
        }

        // scores for 16q x 32k
        f32x8 s0 = {}, s1 = {};
        s0 = __builtin_amdgcn_wmma_f32_16x16x32_bf16(
            false, qa[0].v, false, kf[0][0].v, (short)0, s0, false, false);
        s0 = __builtin_amdgcn_wmma_f32_16x16x32_bf16(
            false, qa[1].v, false, kf[0][1].v, (short)0, s0, false, false);
        s1 = __builtin_amdgcn_wmma_f32_16x16x32_bf16(
            false, qa[0].v, false, kf[1][0].v, (short)0, s1, false, false);
        s1 = __builtin_amdgcn_wmma_f32_16x16x32_bf16(
            false, qa[1].v, false, kf[1][1].v, (short)0, s1, false, false);

        // online softmax over 32 keys; row m = v + 8*half lives in one half
        float corr[8];
#pragma unroll
        for (int v = 0; v < 8; ++v) {
            int i  = i0 + v + 8 * half;
            int jA = j0 + lidx;
            int jB = j0 + 16 + lidx;
            float sA = s0[v] * scale;
            float sB = s1[v] * scale;
            bool mA = (jA > i) || (i - jA > WINH);
            bool mB = (jB > i) || (i - jB > WINH);
            if (mA) sA = -INFINITY;
            if (mB) sB = -INFINITY;

            float tmax = fmaxf(sA, sB);
#pragma unroll
            for (int off = 1; off < 16; off <<= 1)
                tmax = fmaxf(tmax, __shfl_xor(tmax, off, 32));

            float nm = fmaxf(m_i[v], tmax);
            float cr = (nm == -INFINITY) ? 1.0f : __expf(m_i[v] - nm);
            float pA = mA ? 0.0f : __expf(sA - nm);
            float pB = mB ? 0.0f : __expf(sB - nm);

            float rs = pA + pB;
#pragma unroll
            for (int off = 1; off < 16; off <<= 1)
                rs += __shfl_xor(rs, off, 32);

            l_i[v] = l_i[v] * cr + rs;
            m_i[v] = nm;
            corr[v] = cr;
            int row = v + 8 * half;
            pslice[row * 32 + lidx]      = (__bf16)pA;
            pslice[row * 32 + 16 + lidx] = (__bf16)pB;
        }
#pragma unroll
        for (int c = 0; c < 4; ++c)
#pragma unroll
            for (int v = 0; v < 8; ++v) o_acc[c][v] *= corr[v];

        // same-wave DS ops are in-order; fence compiler + wait stores retired
        asm volatile("s_wait_dscnt 0x0" ::: "memory");

        // P as full 16x32 A fragment (no zero padding)
        ABFrag pa;
        pa.h[0] = *(const bf16x8*)&pslice[lidx * 32 + 16 * half];
        pa.h[1] = *(const bf16x8*)&pslice[lidx * 32 + 16 * half + 8];

        // V as 32x16 B fragments: col = d, K = key = e + 16*half (all valid)
        const __bf16* vbase = vb + (((size_t)b * S_ + j0) * KV_ + kvh) * D_;
#pragma unroll
        for (int c2 = 0; c2 < 4; ++c2) {
            ABFrag vf;
#pragma unroll
            for (int e = 0; e < 16; ++e)
                vf.e[e] = vbase[(size_t)(e + 16 * half) * (KV_ * D_)
                                + c2 * 16 + lidx];
            o_acc[c2] = __builtin_amdgcn_wmma_f32_16x16x32_bf16(
                false, pa.v, false, vf.v, (short)0, o_acc[c2], false, false);
        }
        asm volatile("s_wait_dscnt 0x0" ::: "memory");
    }

    // normalize and store: O row q = v+8*half, col d = c2*16+lidx
#pragma unroll
    for (int v = 0; v < 8; ++v) {
        float rl = (l_i[v] > 0.0f) ? (1.0f / l_i[v]) : 0.0f;
        int q = i0 + v + 8 * half;
        __bf16* orow = ob + (((size_t)b * S_ + q) * H_ + h) * D_;
#pragma unroll
        for (int c2 = 0; c2 < 4; ++c2)
            orow[c2 * 16 + lidx] = (__bf16)(o_acc[c2][v] * rl);
    }
}

// ---------------- host orchestration ----------------
extern "C" void kernel_launch(void* const* d_in, const int* in_sizes, int n_in,
                              void* d_out, int out_size, void* d_ws, size_t ws_size,
                              hipStream_t stream) {
    const float* x  = (const float*)d_in[0];
    const float* wq = (const float*)d_in[1];
    const float* wk = (const float*)d_in[2];
    const float* wv = (const float*)d_in[3];
    const float* wo = (const float*)d_in[4];
    float* out = (float*)d_out;

    char* ws = (char*)d_ws;
    size_t off = 0;
    auto alloc_bf = [&](size_t elems) -> __bf16* {
        __bf16* p = (__bf16*)(ws + off);
        off += ((elems * sizeof(__bf16)) + 255) & ~(size_t)255;
        return p;
    };

    __bf16* x_bf  = alloc_bf((size_t)NT_ * DIM_);
    __bf16* wq_bf = alloc_bf((size_t)(H_ * D_) * DIM_);
    __bf16* wk_bf = alloc_bf((size_t)(KV_ * D_) * DIM_);
    __bf16* wv_bf = alloc_bf((size_t)(KV_ * D_) * DIM_);
    __bf16* wo_bf = alloc_bf((size_t)DIM_ * (H_ * D_));
    __bf16* q_bf  = alloc_bf((size_t)NT_ * H_ * D_);
    __bf16* k_bf  = alloc_bf((size_t)NT_ * KV_ * D_);
    __bf16* v_bf  = alloc_bf((size_t)NT_ * KV_ * D_);
    __bf16* a_bf  = alloc_bf((size_t)NT_ * H_ * D_);

    // 1) fp32 -> bf16
    auto cvt = [&](const float* src, __bf16* dst, int n) {
        f32_to_bf16_kernel<<<(n + 255) / 256, 256, 0, stream>>>(src, dst, n);
    };
    cvt(x,  x_bf,  NT_ * DIM_);
    cvt(wq, wq_bf, H_ * D_ * DIM_);
    cvt(wk, wk_bf, KV_ * D_ * DIM_);
    cvt(wv, wv_bf, KV_ * D_ * DIM_);
    cvt(wo, wo_bf, DIM_ * H_ * D_);

    // 2) QKV projections (C = X * W^T), 32x64 macro-tiles, 8 waves/block
    {
        int waves = (NT_ / 32) * ((H_ * D_) / 64);   // 4096
        gemm_nt_bf16_kernel<false><<<waves / 8, 256, 0, stream>>>(
            x_bf, wq_bf, q_bf, NT_, H_ * D_, DIM_);
    }
    {
        int waves = (NT_ / 32) * ((KV_ * D_) / 64);  // 1024
        gemm_nt_bf16_kernel<false><<<waves / 8, 256, 0, stream>>>(
            x_bf, wk_bf, k_bf, NT_, KV_ * D_, DIM_);
        gemm_nt_bf16_kernel<false><<<waves / 8, 256, 0, stream>>>(
            x_bf, wv_bf, v_bf, NT_, KV_ * D_, DIM_);
    }

    // 3) RoPE on Q and K
    {
        int totq = NT_ * H_ * 32;
        rope_kernel<<<(totq + 255) / 256, 256, 0, stream>>>(q_bf, H_, totq);
        int totk = NT_ * KV_ * 32;
        rope_kernel<<<(totk + 255) / 256, 256, 0, stream>>>(k_bf, KV_, totk);
    }

    // 4) sliding-window attention: 8 query tiles per block
    {
        dim3 grid((S_ / 16) / 8, H_, B_);
        attn_kernel<<<grid, 256, 0, stream>>>(q_bf, k_bf, v_bf, a_bf);
    }

    // 5) output projection (fp32 store into d_out)
    {
        int waves = (NT_ / 32) * (DIM_ / 64);        // 4096
        gemm_nt_bf16_kernel<true><<<waves / 8, 256, 0, stream>>>(
            a_bf, wo_bf, out, NT_, DIM_, DIM_);
    }
}
